// GNNLayer_41626823033095
// MI455X (gfx1250) — matrix operand order repaired and checked by reading, not compiled, not run on hardware
//
#include <hip/hip_runtime.h>
#include <hip/hip_bf16.h>
#include <cstdint>
#include <cstddef>

// ---------------------------------------------------------------------------
// CDNA5 (gfx1250) GNN layer: 6 bf16 WMMA GEMMs, BN folded into weights/bias,
// bf16 A operands staged through double-buffered LDS, fused act+stat epilogues.
// ---------------------------------------------------------------------------

typedef __attribute__((ext_vector_type(16))) __bf16       v16bf;
typedef __attribute__((ext_vector_type(2)))  __bf16       v2bf;
typedef __attribute__((ext_vector_type(8)))  float        v8f;
typedef __attribute__((ext_vector_type(8)))  unsigned int v8u;
typedef __attribute__((ext_vector_type(4)))  unsigned int v4u;

#define BN_EPS 1e-5f

#if defined(__has_builtin)
#if __has_builtin(__builtin_amdgcn_cvt_pk_bf16_f32)
#define HAVE_PK_BF16 1
#endif
#endif

__device__ __forceinline__ unsigned short f2bf(float f) {
    unsigned int u = __builtin_bit_cast(unsigned int, f);
    u += 0x7FFFu + ((u >> 16) & 1u);  // round-to-nearest-even
    return (unsigned short)(u >> 16);
}
__device__ __forceinline__ unsigned int f2bf_pk(float a, float b) {
#ifdef HAVE_PK_BF16
    v2bf r = __builtin_amdgcn_cvt_pk_bf16_f32(a, b);
    return __builtin_bit_cast(unsigned int, r);
#else
    return (unsigned int)f2bf(a) | ((unsigned int)f2bf(b) << 16);
#endif
}
__device__ __forceinline__ float bf2f(unsigned short h) {
    unsigned int u = ((unsigned int)h) << 16;
    return __builtin_bit_cast(float, u);
}
__device__ __forceinline__ float sigmoid_f(float x) {
    return 1.0f / (1.0f + __expf(-x));
}

enum { SRC_DIRECT = 0, SRC_GATHER = 1, SRC_CONCAT = 2 };
enum { EPI_RELU_STATS = 0, EPI_SIG_STATS = 1, EPI_SIG_BF16 = 2, EPI_SIG_F32 = 3 };

// ---------------------------------------------------------------------------
// GEMM: C[M,N] = act( A[M,K](bf16) @ Wp[K,N](bf16, BN-folded) + bias ).
// Block: 256 threads = 8 waves (4 row-groups x 2 col-groups). Tile 64 x 256.
// Wave: 16 rows x 128 cols = 8 WMMAs per 32-wide k-step. Double-buffered LDS.
// Wp packed: Wp[((k/32)*N + n)*32 + (k%32)]  (B fragment = 16 contig halves).
// ---------------------------------------------------------------------------
template <int SRC, int EPI>
__global__ __launch_bounds__(256) void gemm_bn_kernel(
    const unsigned short* __restrict__ abf,   // A (bf16): DIRECT/GATHER base, CONCAT lo half
    const unsigned short* __restrict__ abf2,  // CONCAT hi half
    const unsigned short* __restrict__ Wp,    // packed bf16 weights (scale-folded)
    const float* __restrict__ bias,           // folded bias [N]
    unsigned short* __restrict__ out_bf,      // bf16 output (EPI 0..2)
    float* __restrict__ out_f,                // fp32 output (EPI 3)
    float* __restrict__ csum, float* __restrict__ csq,  // column stats (EPI 0..1)
    int N, int K) {
    constexpr bool HAS_STATS = (EPI <= EPI_SIG_STATS);

    __shared__ __align__(16) unsigned short As[2][64][40];  // 64 rows x 32 k, pitch 40
    __shared__ float ssum[256];
    __shared__ float ssq[256];

    const int t      = threadIdx.x;
    const int lane   = t & 31;
    const int wave   = t >> 5;
    const int rowgrp = wave >> 1;
    const int colgrp = wave & 1;
    const int half   = lane >> 4;
    const int l16    = lane & 15;
    const int n0     = blockIdx.x * 256;
    const int m0     = blockIdx.y * 64;

    v8f acc[8] = {};

    // cooperative A staging: thread -> (row, 8-wide k segment); pure b128 copy
    const int arow = t >> 2;
    const int kofs = (t & 3) << 3;  // 0,8,16,24
    const int grow = m0 + arow;

    size_t gbase_s = 0, gbase_d = 0;
    if (SRC == SRC_GATHER) {
        int mb = grow / 12;
        int p  = grow - mb * 12;
        int i  = p / 3;
        int jj = p - i * 3;
        int j  = jj + (jj >= i ? 1 : 0);
        gbase_s = ((size_t)mb * 4 + i) << 8;
        gbase_d = ((size_t)mb * 4 + j) << 8;
    }

    auto stage = [&](int kb, int buf) {
        const int c = (kb << 5) + kofs;
        v4u d;
        if (SRC == SRC_DIRECT) {
            d = *(const v4u*)(abf + (size_t)grow * K + c);
        } else if (SRC == SRC_GATHER) {
            d = *(const v4u*)(abf + ((c & 256) ? gbase_d : gbase_s) + (c & 255));
        } else {  // SRC_CONCAT
            d = (c & 256) ? *(const v4u*)(abf2 + (((size_t)grow) << 8) + (c - 256))
                          : *(const v4u*)(abf + (((size_t)grow) << 8) + c);
        }
        *(v4u*)&As[buf][arow][kofs] = d;
    };

    auto compute = [&](int kb, int buf) {
        v8u araw;
#pragma unroll
        for (int r = 0; r < 8; ++r) {
            int kbase = (r << 1) + (half << 4);
            araw[r] = *(const unsigned int*)&As[buf][rowgrp * 16 + l16][kbase];
        }
        v16bf afrag = __builtin_bit_cast(v16bf, araw);
        const unsigned short* wrow = Wp + ((size_t)kb * N) * 32 + half * 16;
#pragma unroll
        for (int ts = 0; ts < 8; ++ts) {
            int ncol = n0 + colgrp * 128 + ts * 16 + l16;
            v8u braw = *(const v8u*)(wrow + (size_t)ncol * 32);
            acc[ts] = __builtin_amdgcn_wmma_f32_16x16x32_bf16(
                false, afrag, false, __builtin_bit_cast(v16bf, braw),
                (short)0, acc[ts], false, false);
        }
    };

    const int KT = K >> 5;
    stage(0, 0);
    int buf = 0;
    for (int kb = 0; kb < KT; ++kb) {
        __syncthreads();
        if (kb + 1 < KT) stage(kb + 1, buf ^ 1);
        compute(kb, buf);
        buf ^= 1;
    }

    // ---- epilogue ----
    if (HAS_STATS) {
        ssum[t] = 0.f;
        ssq[t]  = 0.f;
        __syncthreads();
    }
#pragma unroll
    for (int ts = 0; ts < 8; ++ts) {
        int   bcol = colgrp * 128 + ts * 16 + l16;  // column within block tile
        int   col  = n0 + bcol;
        float bv   = bias[col];
        float lsum = 0.f, lsq = 0.f;
#pragma unroll
        for (int r = 0; r < 8; ++r) {
            int   row = m0 + rowgrp * 16 + r + half * 8;  // ISA C/D 16x16 layout
            float v   = acc[ts][r] + bv;
            if (EPI == EPI_RELU_STATS) {
                v = fmaxf(v, 0.f);
                out_bf[(size_t)row * N + col] = f2bf(v);
            } else if (EPI == EPI_SIG_F32) {
                v = sigmoid_f(v);
                out_f[(size_t)row * N + col] = v;
            } else {
                v = sigmoid_f(v);
                out_bf[(size_t)row * N + col] = f2bf(v);
            }
            lsum += v;
            lsq  += v * v;
        }
        if (HAS_STATS) {
            atomicAdd(&ssum[bcol], lsum);
            atomicAdd(&ssq[bcol], lsq);
        }
    }
    if (HAS_STATS) {
        __syncthreads();
        atomicAdd(&csum[n0 + t], ssum[t]);
        atomicAdd(&csq[n0 + t], ssq[t]);
    }
}

// ---------------------------------------------------------------------------
// Helper kernels
// ---------------------------------------------------------------------------
__global__ __launch_bounds__(256) void zero_kernel(float* p, int n) {
    int i = blockIdx.x * 256 + threadIdx.x;
    if (i < n) p[i] = 0.f;
}

// fp32 -> bf16 bulk convert (pairs, 4 pairs per thread)
__global__ __launch_bounds__(256) void cvt_bf16_kernel(const float* __restrict__ src,
                                                       unsigned int* __restrict__ dst,
                                                       int npair) {
    int i = (blockIdx.x * 256 + threadIdx.x) * 4;
    if (i + 3 < npair) {
#pragma unroll
        for (int u = 0; u < 4; ++u)
            dst[i + u] = f2bf_pk(src[2 * (i + u)], src[2 * (i + u) + 1]);
    }
}

// pack W[K][N] fp32 -> bf16 fragment layout, scaling row k by s[k] (BN fold)
__global__ __launch_bounds__(256) void pack_kernel(const float* __restrict__ W,
                                                   const float* __restrict__ s,
                                                   unsigned short* __restrict__ Wp,
                                                   int K, int N) {
    int idx = blockIdx.x * 256 + threadIdx.x;
    if (idx >= K * N) return;
    int k = idx / N, n = idx - k * N;
    Wp[(((size_t)(k >> 5)) * N + n) * 32 + (k & 31)] = f2bf(W[(size_t)k * N + n] * s[k]);
}

// folded bias: biasOut[n] = c[n] + sum_k t[k] * W[k][n]   (BN shift fold)
__global__ __launch_bounds__(256) void foldbias_kernel(const float* __restrict__ W,
                                                       const float* __restrict__ tsh,
                                                       const float* __restrict__ c,
                                                       float* __restrict__ biasOut,
                                                       int K, int N) {
    int n = blockIdx.x * 256 + threadIdx.x;
    if (n >= N) return;
    float acc = c[n];
    for (int k = 0; k < K; ++k) acc += tsh[k] * W[(size_t)k * N + n];
    biasOut[n] = acc;
}

// column sum / sumsq of fp32 x [rows x 256]
__global__ __launch_bounds__(256) void colstats256_kernel(const float* __restrict__ x,
                                                          float* __restrict__ csum,
                                                          float* __restrict__ csq) {
    int c  = threadIdx.x;
    int r0 = blockIdx.x * 256;
    float ls = 0.f, lq = 0.f;
    for (int r = r0; r < r0 + 256; ++r) {
        float v = x[((size_t)r << 8) + c];
        ls += v;
        lq += v * v;
    }
    atomicAdd(&csum[c], ls);
    atomicAdd(&csq[c], lq);
}

// x_agg[r][c] = mean over 3 consecutive x_msg rows (bf16 in/out) + col stats
__global__ __launch_bounds__(256) void agg_kernel(const unsigned short* __restrict__ xmsg,
                                                  unsigned short* __restrict__ xagg,
                                                  float* __restrict__ csum,
                                                  float* __restrict__ csq) {
    int c  = threadIdx.x;
    int r0 = blockIdx.x * 64;
    float ls = 0.f, lq = 0.f;
    for (int r = r0; r < r0 + 64; ++r) {
        size_t base = (size_t)r * 768;
        float v = (bf2f(xmsg[base + c]) + bf2f(xmsg[base + 256 + c]) +
                   bf2f(xmsg[base + 512 + c])) * (1.f / 3.f);
        xagg[((size_t)r << 8) + c] = f2bf(v);
        ls += v;
        lq += v * v;
    }
    atomicAdd(&csum[c], ls);
    atomicAdd(&csq[c], lq);
}

// (sum,sumsq) -> folded BN affine: s = g*rsqrt(var+eps), t = b - mu*s
__global__ __launch_bounds__(256) void finalize_kernel(const float* __restrict__ csum,
                                                       const float* __restrict__ csq,
                                                       const float* __restrict__ g,
                                                       const float* __restrict__ b,
                                                       float* __restrict__ scale,
                                                       float* __restrict__ shift,
                                                       int n, int statmod, float invcnt) {
    int i = blockIdx.x * 256 + threadIdx.x;
    if (i >= n) return;
    int   si  = i % statmod;
    float mu  = csum[si] * invcnt;
    float var = csq[si] * invcnt - mu * mu;
    float s   = g[i] * rsqrtf(var + BN_EPS);
    scale[i] = s;
    shift[i] = b[i] - mu * s;
}

// ---------------------------------------------------------------------------
// Launcher
// ---------------------------------------------------------------------------
extern "C" void kernel_launch(void* const* d_in, const int* in_sizes, int n_in,
                              void* d_out, int out_size, void* d_ws, size_t ws_size,
                              hipStream_t stream) {
    (void)in_sizes; (void)n_in; (void)out_size; (void)ws_size;
    const int ROWS  = 16384 * 4;   // 65536
    const int MROWS = 16384 * 12;  // 196608

    const float* x   = (const float*)d_in[0];
    const float* ng0 = (const float*)d_in[1];
    const float* nb0 = (const float*)d_in[2];
    const float* nW0 = (const float*)d_in[3];
    const float* nc0 = (const float*)d_in[4];
    const float* ng1 = (const float*)d_in[5];
    const float* nb1 = (const float*)d_in[6];
    const float* nW1 = (const float*)d_in[7];
    const float* nc1 = (const float*)d_in[8];
    const float* mg0 = (const float*)d_in[9];
    const float* mb0 = (const float*)d_in[10];
    const float* mW0 = (const float*)d_in[11];
    const float* mc0 = (const float*)d_in[12];
    const float* mg1 = (const float*)d_in[13];
    const float* mb1 = (const float*)d_in[14];
    const float* mW1 = (const float*)d_in[15];
    const float* mc1 = (const float*)d_in[16];
    const float* fg0 = (const float*)d_in[17];
    const float* fb0 = (const float*)d_in[18];
    const float* fW0 = (const float*)d_in[19];
    const float* fc0 = (const float*)d_in[20];
    const float* fg1 = (const float*)d_in[21];
    const float* fb1 = (const float*)d_in[22];
    const float* fW1 = (const float*)d_in[23];
    const float* fc1 = (const float*)d_in[24];

    char*  ws  = (char*)d_ws;
    size_t off = 0;
    auto take = [&](size_t bytes) {
        size_t r = off;
        off += (bytes + 255) & ~(size_t)255;
        return r;
    };
    unsigned short* Wn0p = (unsigned short*)(ws + take((size_t)256 * 512 * 2));
    unsigned short* Wn1p = (unsigned short*)(ws + take((size_t)512 * 256 * 2));
    unsigned short* Wm0p = (unsigned short*)(ws + take((size_t)512 * 512 * 2));
    unsigned short* Wm1p = (unsigned short*)(ws + take((size_t)512 * 256 * 2));
    unsigned short* Wf0p = (unsigned short*)(ws + take((size_t)512 * 512 * 2));
    unsigned short* Wf1p = (unsigned short*)(ws + take((size_t)512 * 256 * 2));

    float* stats = (float*)(ws + take((size_t)4608 * 4));
    float* csum_x  = stats + 0;    float* csq_x  = stats + 256;
    float* csum_n1 = stats + 512;  float* csq_n1 = stats + 1024;
    float* csum_m1 = stats + 1536; float* csq_m1 = stats + 2048;
    float* csum_f0 = stats + 2560; float* csq_f0 = stats + 3072;
    float* csum_f1 = stats + 3584; float* csq_f1 = stats + 4096;

    float* ssb = (float*)(ws + take((size_t)5632 * 4));
    float* sN0 = ssb + 0;    float* tN0 = ssb + 256;
    float* sM0 = ssb + 512;  float* tM0 = ssb + 1024;
    float* sN1 = ssb + 1536; float* tN1 = ssb + 2048;
    float* sM1 = ssb + 2560; float* tM1 = ssb + 3072;
    float* sF0 = ssb + 3584; float* tF0 = ssb + 4096;
    float* sF1 = ssb + 4608; float* tF1 = ssb + 5120;

    float* biases = (float*)(ws + take((size_t)2304 * 4));
    float* bN0 = biases + 0;     // 512
    float* bN1 = biases + 512;   // 256
    float* bM0 = biases + 768;   // 512
    float* bM1 = biases + 1280;  // 256
    float* bF0 = biases + 1536;  // 512
    float* bF1 = biases + 2048;  // 256

    unsigned short* xbf  = (unsigned short*)(ws + take((size_t)ROWS * 256 * 2));
    unsigned short* hn   = (unsigned short*)(ws + take((size_t)ROWS * 512 * 2));
    unsigned short* hm   = (unsigned short*)(ws + take((size_t)MROWS * 512 * 2));
    unsigned short* xnod = (unsigned short*)(ws + take((size_t)ROWS * 256 * 2));
    unsigned short* xmsg = (unsigned short*)(ws + take((size_t)MROWS * 256 * 2));
    unsigned short* xagg = (unsigned short*)(ws + take((size_t)ROWS * 256 * 2));
    unsigned short* hf   = (unsigned short*)(ws + take((size_t)ROWS * 512 * 2));

    // 0) zero stat accumulators (deterministic every launch)
    zero_kernel<<<(4608 + 255) / 256, 256, 0, stream>>>(stats, 4608);

    // 1) x -> bf16 once; column stats of x (serve node BN0 AND message BN0:
    //    each node appears exactly 3x as src and 3x as dst => identical stats)
    cvt_bf16_kernel<<<(ROWS * 256 / 2 / 4 + 255) / 256, 256, 0, stream>>>(
        x, (unsigned int*)xbf, ROWS * 256 / 2);
    colstats256_kernel<<<ROWS / 256, 256, 0, stream>>>(x, csum_x, csq_x);
    finalize_kernel<<<1, 256, 0, stream>>>(csum_x, csq_x, ng0, nb0, sN0, tN0, 256, 256, 1.f / ROWS);
    finalize_kernel<<<2, 256, 0, stream>>>(csum_x, csq_x, mg0, mb0, sM0, tM0, 512, 256, 1.f / ROWS);

    // 2) node MLP
    pack_kernel<<<(256 * 512 + 255) / 256, 256, 0, stream>>>(nW0, sN0, Wn0p, 256, 512);
    foldbias_kernel<<<2, 256, 0, stream>>>(nW0, tN0, nc0, bN0, 256, 512);
    gemm_bn_kernel<SRC_DIRECT, EPI_RELU_STATS><<<dim3(2, ROWS / 64), 256, 0, stream>>>(
        xbf, nullptr, Wn0p, bN0, hn, nullptr, csum_n1, csq_n1, 512, 256);
    finalize_kernel<<<2, 256, 0, stream>>>(csum_n1, csq_n1, ng1, nb1, sN1, tN1, 512, 512, 1.f / ROWS);
    pack_kernel<<<(512 * 256 + 255) / 256, 256, 0, stream>>>(nW1, sN1, Wn1p, 512, 256);
    foldbias_kernel<<<1, 256, 0, stream>>>(nW1, tN1, nc1, bN1, 512, 256);
    gemm_bn_kernel<SRC_DIRECT, EPI_SIG_STATS><<<dim3(1, ROWS / 64), 256, 0, stream>>>(
        hn, nullptr, Wn1p, bN1, xnod, nullptr, csum_f0, csq_f0, 256, 512);

    // 3) message MLP (A rows gathered from xbf on the fly; msgs never built)
    pack_kernel<<<(512 * 512 + 255) / 256, 256, 0, stream>>>(mW0, sM0, Wm0p, 512, 512);
    foldbias_kernel<<<2, 256, 0, stream>>>(mW0, tM0, mc0, bM0, 512, 512);
    gemm_bn_kernel<SRC_GATHER, EPI_RELU_STATS><<<dim3(2, MROWS / 64), 256, 0, stream>>>(
        xbf, nullptr, Wm0p, bM0, hm, nullptr, csum_m1, csq_m1, 512, 512);
    finalize_kernel<<<2, 256, 0, stream>>>(csum_m1, csq_m1, mg1, mb1, sM1, tM1, 512, 512, 1.f / MROWS);
    pack_kernel<<<(512 * 256 + 255) / 256, 256, 0, stream>>>(mW1, sM1, Wm1p, 512, 256);
    foldbias_kernel<<<1, 256, 0, stream>>>(mW1, tM1, mc1, bM1, 512, 256);
    gemm_bn_kernel<SRC_DIRECT, EPI_SIG_BF16><<<dim3(1, MROWS / 64), 256, 0, stream>>>(
        hm, nullptr, Wm1p, bM1, xmsg, nullptr, nullptr, nullptr, 256, 512);

    // 4) aggregate (mean over 3) + x_agg half of final BN0 stats
    agg_kernel<<<ROWS / 64, 256, 0, stream>>>(xmsg, xagg, csum_f0 + 256, csq_f0 + 256);
    finalize_kernel<<<2, 256, 0, stream>>>(csum_f0, csq_f0, fg0, fb0, sF0, tF0, 512, 512, 1.f / ROWS);

    // 5) final MLP on concat(x_node, x_agg)
    pack_kernel<<<(512 * 512 + 255) / 256, 256, 0, stream>>>(fW0, sF0, Wf0p, 512, 512);
    foldbias_kernel<<<2, 256, 0, stream>>>(fW0, tF0, fc0, bF0, 512, 512);
    gemm_bn_kernel<SRC_CONCAT, EPI_RELU_STATS><<<dim3(2, ROWS / 64), 256, 0, stream>>>(
        xnod, xagg, Wf0p, bF0, hf, nullptr, csum_f1, csq_f1, 512, 512);
    finalize_kernel<<<2, 256, 0, stream>>>(csum_f1, csq_f1, fg1, fb1, sF1, tF1, 512, 512, 1.f / ROWS);
    pack_kernel<<<(512 * 256 + 255) / 256, 256, 0, stream>>>(fW1, sF1, Wf1p, 512, 256);
    foldbias_kernel<<<1, 256, 0, stream>>>(fW1, tF1, fc1, bF1, 512, 256);
    gemm_bn_kernel<SRC_DIRECT, EPI_SIG_F32><<<dim3(1, ROWS / 64), 256, 0, stream>>>(
        hf, nullptr, Wf1p, bF1, nullptr, (float*)d_out, nullptr, nullptr, 256, 512);
}
